// BiMamba2_35012573397401
// MI455X (gfx1250) — matrix operand order repaired and checked by reading, not compiled, not run on hardware
//
#include <hip/hip_runtime.h>
#include <math.h>

// ---------------- model dims ----------------
#define BSZ      2
#define LSEQ     4096
#define DMODEL   1024
#define DINNER   2048
#define HEADDIM  64
#define NHEADS   32
#define DSTATE   128
#define DCONV    4
#define CHUNK    256
#define NCHUNK   (LSEQ / CHUNK)        // 16
#define DPROJ    (2*DINNER + 2*DSTATE + NHEADS)   // 4384
#define CONVDIM  (DINNER + 2*DSTATE)              // 2304
#define BOFF     DINNER                // B cols inside xBC
#define COFF     (DINNER + DSTATE)     // C cols inside xBC

typedef __attribute__((ext_vector_type(16))) __bf16 v16bf;
typedef __attribute__((ext_vector_type(8)))  float  v8f;
typedef __attribute__((ext_vector_type(4))) unsigned v4u;
typedef __attribute__((ext_vector_type(8)))  int     v8i;
typedef __attribute__((ext_vector_type(4)))  int     v4i;

#if defined(__has_builtin)
#  if __has_builtin(__builtin_amdgcn_tensor_load_to_lds)
#    define HAVE_TDM 1
#  endif
#endif
#ifndef HAVE_TDM
#  define HAVE_TDM 0
#endif

// ---------------- CDNA5 async global->LDS copy (ASYNCcnt) ----------------
__device__ __forceinline__ void async_copy16(void* lds, const void* g) {
  unsigned l = (unsigned)(unsigned long long)lds;   // low 32 bits = LDS byte address
  asm volatile("global_load_async_to_lds_b128 %0, %1, off"
               :: "v"(l), "v"(g) : "memory");
}
#define ASYNC_WAIT(N) asm volatile("s_wait_asynccnt %0" :: "n"(N) : "memory")

#if HAVE_TDM
// ---------------- CDNA5 Tensor Data Mover: 2D bf16 tile -> LDS ----------------
// D# per CDNA5 ISA: group0 {count,lds,global,type},
// group1 {data_size, tensor_dim0/1, tile_dim0/1, dim0_stride}. 2D => groups 2/3 zero.
__device__ __forceinline__ void tdm_load_2d(
    unsigned lds_addr, const void* gaddr,
    unsigned tensor_d0, unsigned tensor_d1,     // tensor extent (elements, rows)
    unsigned long long d0_stride,               // row stride (elements)
    unsigned tile_d0, unsigned tile_d1)         // tile extent (elements, rows)
{
  unsigned long long ga = (unsigned long long)gaddr;
  v4u g0;
  g0[0] = 1u;                                        // count=1, user-mode, no gather
  g0[1] = lds_addr;                                  // bits 63:32  lds_addr
  g0[2] = (unsigned)(ga & 0xffffffffu);              // bits 95:64  global_addr lo
  g0[3] = (unsigned)((ga >> 32) & 0x01ffffffu) | (2u << 30); // addr hi | type=2
  v8i g1;
  g1[0] = (int)(1u << 16);                           // data_size=1 (2 bytes/elem)
  g1[1] = (int)((tensor_d0 & 0xffffu) << 16);        // tensor_dim0[15:0] @bit48
  g1[2] = (int)((tensor_d0 >> 16) | ((tensor_d1 & 0xffffu) << 16)); // dim0 hi | dim1 lo
  g1[3] = (int)((tensor_d1 >> 16) | (tile_d0 << 16)); // dim1 hi | tile_dim0
  g1[4] = (int)(tile_d1 & 0xffffu);                  // tile_dim1 ; tile_dim2=0
  g1[5] = (int)(unsigned)(d0_stride & 0xffffffffu);  // dim0_stride[31:0]
  g1[6] = (int)(unsigned)((d0_stride >> 32) & 0xffffu); // dim0_stride[47:32]
  g1[7] = 0;
  v4i z4 = {0, 0, 0, 0};
#if __clang_major__ >= 23
  v8i z8 = {0, 0, 0, 0, 0, 0, 0, 0};
  __builtin_amdgcn_tensor_load_to_lds(g0, g1, z4, z4, z8, 0);
#else
  __builtin_amdgcn_tensor_load_to_lds(g0, g1, z4, z4, 0);
#endif
}
#endif

// ---------------- WMMA helpers (CDNA5 16x16x32 bf16, f32 accum) ----------------
__device__ __forceinline__ v8f wmma_bf16(v16bf a, v16bf b, v8f c) {
  return __builtin_amdgcn_wmma_f32_16x16x32_bf16(false, a, false, b, (short)0, c, false, false);
}
__device__ __forceinline__ v8f zero8() {
  v8f z;
#pragma unroll
  for (int i = 0; i < 8; ++i) z[i] = 0.f;
  return z;
}
__device__ __forceinline__ int a_k_of(int j, int half) {
  return half*16 + 2*j;
}
__device__ __forceinline__ v16bf lds_frag_a(const __bf16* p, int stride) {
  const int l = threadIdx.x & 31, r = l & 15, hf = l >> 4;
  v16bf f;
#pragma unroll
  for (int j = 0; j < 8; ++j) {
    int k = a_k_of(j, hf);
    f[2*j]   = p[r*stride + k];
    f[2*j+1] = p[r*stride + k + 1];
  }
  return f;
}

// ---------------- 0) f32 -> bf16 bulk convert ----------------
__global__ __launch_bounds__(256) void k_cvt(
    const float* __restrict__ s, __bf16* __restrict__ d, long long n)
{
  long long i = ((long long)blockIdx.x*256 + threadIdx.x)*8;
  if (i + 8 <= n) {
    float4 a = *(const float4*)(s + i);
    float4 b = *(const float4*)(s + i + 4);
    __bf16 o[8] = {(__bf16)a.x,(__bf16)a.y,(__bf16)a.z,(__bf16)a.w,
                   (__bf16)b.x,(__bf16)b.y,(__bf16)b.z,(__bf16)b.w};
    *(float4*)(d + i) = *(const float4*)o;
  }
}

// bi-dir merge to bf16: dst[b,l,:] = bf16(0.5*(y0[b,l,:] + y1[b,L-1-l,:]))
__global__ __launch_bounds__(256) void k_combine(
    const float* __restrict__ y0, const float* __restrict__ y1,
    __bf16* __restrict__ d)
{
  long long idx = (long long)blockIdx.x*256 + threadIdx.x;
  long long e8 = idx*8;
  int m  = (int)(e8 / DINNER);
  int k0 = (int)(e8 % DINNER);
  int b = m >> 12, l = m & (LSEQ - 1);
  const float* p0 = y0 + (size_t)m*DINNER + k0;
  const float* p1 = y1 + (size_t)(b*LSEQ + (LSEQ-1-l))*DINNER + k0;
  float4 a0 = *(const float4*)p0, a1 = *(const float4*)(p0+4);
  float4 b0 = *(const float4*)p1, b1 = *(const float4*)(p1+4);
  __bf16 o[8] = {(__bf16)(0.5f*(a0.x+b0.x)), (__bf16)(0.5f*(a0.y+b0.y)),
                 (__bf16)(0.5f*(a0.z+b0.z)), (__bf16)(0.5f*(a0.w+b0.w)),
                 (__bf16)(0.5f*(a1.x+b1.x)), (__bf16)(0.5f*(a1.y+b1.y)),
                 (__bf16)(0.5f*(a1.z+b1.z)), (__bf16)(0.5f*(a1.w+b1.w))};
  *(float4*)(d + e8) = *(const float4*)o;
}

// ---------------- 1) bf16 GEMM: C[M,N] = A[M,K] . W[N,K]^T ----------------
// double-buffered TDM (fallback: async-to-LDS) staging; 128x128 block, 8 waves
__global__ __launch_bounds__(256) void k_gemm_bf16(
    const __bf16* __restrict__ A, const __bf16* __restrict__ W,
    float* __restrict__ C, int M, int N, int K)
{
  __shared__ __bf16 As[2][128][32];
  __shared__ __bf16 Bs[2][128][32];
  const int tid = threadIdx.x;
  const int m0 = blockIdx.x * 128;
  const int n0 = blockIdx.y * 128;
  const int wave = tid >> 5;
  const int wm = wave >> 2, wn = wave & 3;
  const int lane = tid & 31, cc = lane & 15, hf = lane >> 4;

  v8f acc[4][2];
#pragma unroll
  for (int i = 0; i < 4; ++i)
#pragma unroll
    for (int j = 0; j < 2; ++j) acc[i][j] = zero8();

#if HAVE_TDM
  auto stage = [&](int kt, int buf) {
    if (wave == 0) {   // one wave drives the DMA engine; EXEC-independent issue
      tdm_load_2d((unsigned)(unsigned long long)&As[buf][0][0],
                  A + (size_t)m0*K + kt,
                  (unsigned)(K - kt), (unsigned)(M - m0), (unsigned)K, 32u, 128u);
      tdm_load_2d((unsigned)(unsigned long long)&Bs[buf][0][0],
                  W + (size_t)n0*K + kt,
                  (unsigned)(K - kt), (unsigned)(N - n0), (unsigned)K, 32u, 128u);
    }
  };
#else
  auto stage = [&](int kt, int buf) {
#pragma unroll
    for (int it = 0; it < 2; ++it) {
      int cid = tid + it*256;
      int r = cid >> 2;
      int c8 = (cid & 3) * 8;
      int gr = m0 + r; if (gr >= M) gr = M - 1;
      int gn = n0 + r; if (gn >= N) gn = N - 1;
      async_copy16(&As[buf][r][c8], A + (size_t)gr*K + kt + c8);
      async_copy16(&Bs[buf][r][c8], W + (size_t)gn*K + kt + c8);
    }
  };
#endif

  stage(0, 0);
  const int nk = K / 32;
  for (int ki = 0; ki < nk; ++ki) {
    int buf = ki & 1;
    int pre = (ki + 1 < nk);
    if (pre) stage((ki+1)*32, buf ^ 1);
#if HAVE_TDM
    if (wave == 0) {
      if (pre) __builtin_amdgcn_s_wait_tensorcnt(2);  // tile ki landed
      else     __builtin_amdgcn_s_wait_tensorcnt(0);
    }
#else
    if (pre) { ASYNC_WAIT(4); } else { ASYNC_WAIT(0); }
#endif
    __syncthreads();

    v16bf af[4], bfv[2];
#pragma unroll
    for (int mi = 0; mi < 4; ++mi)
      af[mi] = lds_frag_a(&As[buf][wm*64 + mi*16][0], 32);
#pragma unroll
    for (int ni = 0; ni < 2; ++ni) {
      const __bf16* p = &Bs[buf][wn*32 + ni*16 + cc][0];
      v16bf f;
#pragma unroll
      for (int j = 0; j < 8; ++j) {
        int k = hf*16 + 2*j;
        f[2*j] = p[k]; f[2*j+1] = p[k+1];
      }
      bfv[ni] = f;
    }
#pragma unroll
    for (int mi = 0; mi < 4; ++mi)
#pragma unroll
      for (int ni = 0; ni < 2; ++ni)
        acc[mi][ni] = wmma_bf16(af[mi], bfv[ni], acc[mi][ni]);
    __syncthreads();
  }
#pragma unroll
  for (int mi = 0; mi < 4; ++mi)
#pragma unroll
    for (int ni = 0; ni < 2; ++ni) {
      int col = n0 + wn*32 + ni*16 + cc;
      if (col < N) {
#pragma unroll
        for (int v = 0; v < 8; ++v) {
          int row = m0 + wm*64 + mi*16 + v + 8*hf;
          if (row < M) C[(size_t)row*N + col] = acc[mi][ni][v];
        }
      }
    }
}

// ---------------- 2) depthwise causal conv + SiLU -> bf16 xBC ----------------
__global__ __launch_bounds__(256) void k_conv(
    const float* __restrict__ zx, const float* __restrict__ cw,
    const float* __restrict__ cb, __bf16* __restrict__ xbcb, int rev)
{
  int idx = blockIdx.x*256 + threadIdx.x;
  if (idx >= BSZ*LSEQ*CONVDIM) return;
  int ch = idx % CONVDIM;
  int t  = (idx / CONVDIM) % LSEQ;
  int b  = idx / (CONVDIM*LSEQ);
  float acc = cb[ch];
#pragma unroll
  for (int k = 0; k < DCONV; ++k) {
    int ts = t + k - (DCONV-1);
    if (ts >= 0) {
      int lsrc = rev ? (LSEQ-1-ts) : ts;
      acc += cw[ch*DCONV + k] * zx[(size_t)(b*LSEQ + lsrc)*DPROJ + DINNER + ch];
    }
  }
  xbcb[(size_t)(b*LSEQ + t)*CONVDIM + ch] = (__bf16)(acc / (1.f + __expf(-acc)));
}

// ---------------- 3) dt softplus + per-chunk cumsum of A*dt ----------------
__global__ __launch_bounds__(256) void k_dtcs(
    const float* __restrict__ zx, const float* __restrict__ dt_bias,
    const float* __restrict__ A_log, float* __restrict__ dtb,
    float* __restrict__ acs, int rev)
{
  __shared__ float s[CHUNK];
  int bi = blockIdx.x;
  int c = bi & 15, hh = (bi >> 4) & 31, b = bi >> 9;
  int t = threadIdx.x;
  int l = c*CHUNK + t;
  int lsrc = rev ? (LSEQ-1-l) : l;
  float raw = zx[(size_t)(b*LSEQ + lsrc)*DPROJ + DINNER + CONVDIM + hh] + dt_bias[hh];
  float dt = (raw > 20.f) ? raw : log1pf(__expf(raw));
  dtb[(size_t)(b*NHEADS + hh)*LSEQ + l] = dt;
  float A = -__expf(A_log[hh]);
  s[t] = dt * A;
  __syncthreads();
  for (int off = 1; off < CHUNK; off <<= 1) {
    float v = (t >= off) ? s[t-off] : 0.f;
    __syncthreads();
    s[t] += v;
    __syncthreads();
  }
  acs[(size_t)(b*NHEADS + hh)*LSEQ + l] = s[t];
}

// ---------------- 4) chunk states (double-buffered async staging) ----------------
__global__ __launch_bounds__(256) void k_states(
    const __bf16* __restrict__ xbcb, const float* __restrict__ dtb,
    const float* __restrict__ acs, float* __restrict__ states)
{
  __shared__ float  s_w[CHUNK];
  __shared__ __bf16 Bsl[2][32][128];
  __shared__ __bf16 Xsl[2][32][64];
  int bi = blockIdx.x;
  int c = bi & 15, hh = (bi >> 4) & 31, b = bi >> 9;
  int tid = threadIdx.x;
  const float* acs_p = acs + (size_t)(b*NHEADS + hh)*LSEQ + c*CHUNK;
  const float* dt_p  = dtb + (size_t)(b*NHEADS + hh)*LSEQ + c*CHUNK;
  float a_last = acs_p[CHUNK-1];
  s_w[tid] = dt_p[tid] * __expf(a_last - acs_p[tid]);

  int wave = tid >> 5, lane = tid & 31, r = lane & 15, hf = lane >> 4;
  int wm = wave >> 2, wn = wave & 3;
  const __bf16* xb = xbcb + (size_t)(b*LSEQ + c*CHUNK)*CONVDIM;

  auto stage = [&](int kk, int buf) {
#pragma unroll
    for (int it = 0; it < 2; ++it) {
      int cid = tid + it*256;
      int rr = cid >> 4, c8 = (cid & 15)*8;
      async_copy16(&Bsl[buf][rr][c8], xb + (size_t)(kk*32 + rr)*CONVDIM + BOFF + c8);
    }
    int rr = tid >> 3, c8 = (tid & 7)*8;
    async_copy16(&Xsl[buf][rr][c8], xb + (size_t)(kk*32 + rr)*CONVDIM + hh*HEADDIM + c8);
  };

  v8f acc[2][2];
#pragma unroll
  for (int i = 0; i < 2; ++i)
#pragma unroll
    for (int j = 0; j < 2; ++j) acc[i][j] = zero8();

  stage(0, 0);
  for (int kk = 0; kk < CHUNK/32; ++kk) {
    int buf = kk & 1;
    if (kk + 1 < CHUNK/32) { stage(kk+1, buf ^ 1); ASYNC_WAIT(3); }
    else                   { ASYNC_WAIT(0); }
    __syncthreads();

    v16bf bfv[2];
#pragma unroll
    for (int ni = 0; ni < 2; ++ni) {
      int n = wn*32 + ni*16 + r;
      v16bf f;
#pragma unroll
      for (int j = 0; j < 8; ++j) {
        int tk = hf*16 + 2*j;
        f[2*j]   = Bsl[buf][tk][n];
        f[2*j+1] = Bsl[buf][tk+1][n];
      }
      bfv[ni] = f;
    }
#pragma unroll
    for (int mi = 0; mi < 2; ++mi) {
      int p = wm*32 + mi*16 + r;
      v16bf af;
#pragma unroll
      for (int j = 0; j < 8; ++j) {
        int tk = a_k_of(j, hf);
        af[2*j]   = (__bf16)((float)Xsl[buf][tk][p]   * s_w[kk*32 + tk]);
        af[2*j+1] = (__bf16)((float)Xsl[buf][tk+1][p] * s_w[kk*32 + tk + 1]);
      }
#pragma unroll
      for (int ni = 0; ni < 2; ++ni)
        acc[mi][ni] = wmma_bf16(af, bfv[ni], acc[mi][ni]);
    }
    __syncthreads();
  }
  float* so = states + (size_t)((b*NCHUNK + c)*NHEADS + hh)*HEADDIM*DSTATE;
#pragma unroll
  for (int mi = 0; mi < 2; ++mi)
#pragma unroll
    for (int ni = 0; ni < 2; ++ni)
#pragma unroll
      for (int v = 0; v < 8; ++v) {
        int p = wm*32 + mi*16 + v + 8*hf;
        int n = wn*32 + ni*16 + r;
        so[(size_t)p*DSTATE + n] = acc[mi][ni][v];
      }
}

// ---------------- 5) inter-chunk recurrence -> bf16 prev ----------------
__global__ __launch_bounds__(256) void k_recur(
    const float* __restrict__ states, const float* __restrict__ acs,
    __bf16* __restrict__ prevb)
{
  int bi = blockIdx.x;
  int hh = bi & 31, b = bi >> 5;
  int tid = threadIdx.x;
  for (int e = tid; e < HEADDIM*DSTATE; e += 256) {
    float carry = 0.f;
    for (int c = 0; c < NCHUNK; ++c) {
      size_t idx = (size_t)((b*NCHUNK + c)*NHEADS + hh)*HEADDIM*DSTATE + e;
      prevb[idx] = (__bf16)carry;
      float ssum = acs[(size_t)(b*NHEADS + hh)*LSEQ + c*CHUNK + CHUNK-1];
      carry = carry * __expf(ssum) + states[idx];
    }
  }
}

// ---------------- 6) fused chunk scan: Y = Y_diag + Y_off + x*D ----------------
__global__ __launch_bounds__(256) void k_scan(
    const __bf16* __restrict__ xbcb, const float* __restrict__ dtb,
    const float* __restrict__ acs, const __bf16* __restrict__ prevb,
    const float* __restrict__ Dp, float* __restrict__ yraw)
{
  __shared__ float  s_acs[CHUNK], s_dt[CHUNK];
  __shared__ __bf16 pvs[64][128];
  __shared__ __bf16 Bsl[2][32][128];
  __shared__ __bf16 Xsl[2][32][64];
  __shared__ __bf16 s_g[8][16*32];

  int bi = blockIdx.x;
  int c = bi & 15, hh = (bi >> 4) & 31, b = bi >> 9;
  int tid = threadIdx.x, wave = tid >> 5, lane = tid & 31, r = lane & 15, hf = lane >> 4;
  const float* acs_p = acs + (size_t)(b*NHEADS + hh)*LSEQ + c*CHUNK;
  const float* dt_p  = dtb + (size_t)(b*NHEADS + hh)*LSEQ + c*CHUNK;
  s_acs[tid] = acs_p[tid];
  s_dt[tid]  = dt_p[tid];

  const __bf16* pv = prevb + (size_t)((b*NCHUNK + c)*NHEADS + hh)*HEADDIM*DSTATE;
#pragma unroll
  for (int it = 0; it < 4; ++it) {       // prev 64x128 bf16 = 16KB, overlapped below
    int cid = tid + it*256;
    int rr = cid >> 4, c8 = (cid & 15)*8;
    async_copy16(&pvs[rr][c8], pv + (size_t)rr*DSTATE + c8);
  }

  const __bf16* xb = xbcb + (size_t)(b*LSEQ + c*CHUNK)*CONVDIM;
  float dval = Dp[hh];
  __bf16* gbuf = &s_g[wave][0];

  // C row-tile A fragments (global bf16) — overlaps prev DMA
  v16bf afC[2][4];
#pragma unroll
  for (int ti = 0; ti < 2; ++ti) {
    int t0 = (wave + ti*8)*16;
#pragma unroll
    for (int kk = 0; kk < 4; ++kk) {
      v16bf f;
#pragma unroll
      for (int j = 0; j < 8; ++j) {
        int k = a_k_of(j, hf);
        f[2*j]   = xb[(size_t)(t0+r)*CONVDIM + COFF + kk*32 + k];
        f[2*j+1] = xb[(size_t)(t0+r)*CONVDIM + COFF + kk*32 + k + 1];
      }
      afC[ti][kk] = f;
    }
  }
  ASYNC_WAIT(0);
  __syncthreads();

  auto stage = [&](int sb, int buf) {
#pragma unroll
    for (int it = 0; it < 2; ++it) {
      int cid = tid + it*256;
      int rr = cid >> 4, c8 = (cid & 15)*8;
      async_copy16(&Bsl[buf][rr][c8], xb + (size_t)(sb*32 + rr)*CONVDIM + BOFF + c8);
    }
    int rr = tid >> 3, c8 = (tid & 7)*8;
    async_copy16(&Xsl[buf][rr][c8], xb + (size_t)(sb*32 + rr)*CONVDIM + hh*HEADDIM + c8);
  };
  stage(0, 0);                            // prefetch first s-block, overlaps Y_off

  v8f acc[2][4];
#pragma unroll
  for (int ti = 0; ti < 2; ++ti) {
    int t0 = (wave + ti*8)*16;
#pragma unroll
    for (int ni = 0; ni < 4; ++ni) acc[ti][ni] = zero8();
    // Y_off = exp(acs[t]) * C_t . prev^T  (prev from LDS)
    float rs = __expf(s_acs[t0 + r]);
#pragma unroll
    for (int kk = 0; kk < 4; ++kk) {
      v16bf aS;
#pragma unroll
      for (int e = 0; e < 16; ++e) aS[e] = (__bf16)((float)afC[ti][kk][e] * rs);
#pragma unroll
      for (int ni = 0; ni < 4; ++ni) {
        v16bf f;
#pragma unroll
        for (int j = 0; j < 8; ++j) {
          int n = kk*32 + hf*16 + 2*j;
          f[2*j]   = pvs[ni*16 + r][n];
          f[2*j+1] = pvs[ni*16 + r][n + 1];
        }
        acc[ti][ni] = wmma_bf16(aS, f, acc[ti][ni]);
      }
    }
  }

  // causal diagonal: cooperative s-blocks of 32, double-buffered
  for (int sb = 0; sb < CHUNK/32; ++sb) {
    int buf = sb & 1;
    if (sb + 1 < CHUNK/32) { stage(sb+1, buf ^ 1); ASYNC_WAIT(3); }
    else                   { ASYNC_WAIT(0); }
    __syncthreads();

#pragma unroll
    for (int ti = 0; ti < 2; ++ti) {
      int tt = wave + ti*8;
      if ((tt >> 1) >= sb) {             // wave-uniform causal condition
        int t0 = tt*16;
#pragma unroll
        for (int ss = 0; ss < 2; ++ss) {
          int s0 = sb*32 + ss*16;
          v8f g = zero8();
#pragma unroll
          for (int kk = 0; kk < 4; ++kk) {
            v16bf f;
#pragma unroll
            for (int j = 0; j < 8; ++j) {
              int n = kk*32 + hf*16 + 2*j;
              f[2*j]   = Bsl[buf][ss*16 + r][n];
              f[2*j+1] = Bsl[buf][ss*16 + r][n + 1];
            }
            g = wmma_bf16(afC[ti][kk], f, g);
          }
#pragma unroll
          for (int v = 0; v < 8; ++v) {
            int trow = t0 + v + 8*hf;
            int scol = s0 + r;
            float m = (scol <= trow) ? g[v] * __expf(s_acs[trow] - s_acs[scol]) : 0.f;
            gbuf[(v + 8*hf)*32 + ss*16 + r] = (__bf16)m;
          }
        }
        v16bf am = lds_frag_a(gbuf, 32);
#pragma unroll
        for (int ni = 0; ni < 4; ++ni) {
          v16bf f;
#pragma unroll
          for (int j = 0; j < 8; ++j) {
            int sk = hf*16 + 2*j;
            int p = ni*16 + r;
            f[2*j]   = (__bf16)((float)Xsl[buf][sk][p]   * s_dt[sb*32 + sk]);
            f[2*j+1] = (__bf16)((float)Xsl[buf][sk+1][p] * s_dt[sb*32 + sk + 1]);
          }
          acc[ti][ni] = wmma_bf16(am, f, acc[ti][ni]);
        }
      }
    }
    __syncthreads();
  }

  // store Y + x*D
  float* yo = yraw + (size_t)(b*LSEQ + c*CHUNK)*DINNER + hh*HEADDIM;
#pragma unroll
  for (int ti = 0; ti < 2; ++ti) {
    int t0 = (wave + ti*8)*16;
#pragma unroll
    for (int ni = 0; ni < 4; ++ni)
#pragma unroll
      for (int v = 0; v < 8; ++v) {
        int trow = t0 + v + 8*hf;
        int p = ni*16 + r;
        float xv = (float)xb[(size_t)trow*CONVDIM + hh*HEADDIM + p];
        yo[(size_t)trow*DINNER + p] = acc[ti][ni][v] + xv * dval;
      }
  }
}

// ---------------- 7) SiLU gate + RMSNorm (in place on yraw) ----------------
__global__ __launch_bounds__(256) void k_gnorm(
    const float* __restrict__ zx, const float* __restrict__ normw,
    float* __restrict__ yraw, int rev)
{
  __shared__ float red[256];
  int row = blockIdx.x;
  int b = row / LSEQ, t = row % LSEQ;
  int lsrc = rev ? (LSEQ-1-t) : t;
  const float* zrow = zx + (size_t)(b*LSEQ + lsrc)*DPROJ;
  float* y = yraw + (size_t)row*DINNER;
  int tid = threadIdx.x;
  float vals[DINNER/256];
  float loc = 0.f;
#pragma unroll
  for (int i = 0; i < DINNER/256; ++i) {
    int col = tid + i*256;
    float z = zrow[col];
    float v = y[col] * (z / (1.f + __expf(-z)));
    vals[i] = v;
    loc += v*v;
  }
  red[tid] = loc;
  __syncthreads();
  for (int off = 128; off > 0; off >>= 1) {
    if (tid < off) red[tid] += red[tid + off];
    __syncthreads();
  }
  float scale = rsqrtf(red[0]/(float)DINNER + 1e-5f);
#pragma unroll
  for (int i = 0; i < DINNER/256; ++i) {
    int col = tid + i*256;
    y[col] = vals[i] * scale * normw[col];
  }
}

// ---------------- host-side launcher ----------------
extern "C" void kernel_launch(void* const* d_in, const int* in_sizes, int n_in,
                              void* d_out, int out_size, void* d_ws, size_t ws_size,
                              hipStream_t stream) {
  (void)in_sizes; (void)n_in; (void)out_size; (void)ws_size;
  const float* u      = (const float*)d_in[0];
  const float* ipw    = (const float*)d_in[1];
  const float* cwF    = (const float*)d_in[2];
  const float* cbF    = (const float*)d_in[3];
  const float* dtbF   = (const float*)d_in[4];
  const float* alF    = (const float*)d_in[5];
  const float* DF     = (const float*)d_in[6];
  const float* nwF    = (const float*)d_in[7];
  const float* cwB    = (const float*)d_in[8];
  const float* cbB    = (const float*)d_in[9];
  const float* dtbB   = (const float*)d_in[10];
  const float* alB    = (const float*)d_in[11];
  const float* DB     = (const float*)d_in[12];
  const float* nwB    = (const float*)d_in[13];
  const float* opw    = (const float*)d_in[14];
  float* out = (float*)d_out;

  float* ws = (float*)d_ws;
  size_t off = 0;
  auto alloc_f  = [&](size_t n) { float*  p = ws + off; off += n; return p; };
  auto alloc_bf = [&](size_t n) { __bf16* p = (__bf16*)(ws + off); off += (n + 1) / 2; return p; };

  const int M = BSZ*LSEQ;   // 8192
  float*  zx   = alloc_f((size_t)M*DPROJ);
  __bf16* ubf  = alloc_bf((size_t)M*DMODEL);
  __bf16* ipwb = alloc_bf((size_t)DPROJ*DMODEL);
  __bf16* opwb = alloc_bf((size_t)DMODEL*DINNER);
  __bf16* ycb  = alloc_bf((size_t)M*DINNER);
  __bf16 *xbcb[2], *pvb[2];
  float  *dtb[2], *acsb[2], *st[2], *yr[2];
  for (int d = 0; d < 2; ++d) {
    xbcb[d] = alloc_bf((size_t)M*CONVDIM);
    dtb[d]  = alloc_f((size_t)BSZ*NHEADS*LSEQ);
    acsb[d] = alloc_f((size_t)BSZ*NHEADS*LSEQ);
    st[d]   = alloc_f((size_t)BSZ*NCHUNK*NHEADS*HEADDIM*DSTATE);
    pvb[d]  = alloc_bf((size_t)BSZ*NCHUNK*NHEADS*HEADDIM*DSTATE);
    yr[d]   = alloc_f((size_t)M*DINNER);
  }

  dim3 blk(256);

  // bf16 operand prep
  k_cvt<<<(int)(((size_t)M*DMODEL)/2048),      blk, 0, stream>>>(u,   ubf,  (long long)M*DMODEL);
  k_cvt<<<(int)(((size_t)DPROJ*DMODEL)/2048),  blk, 0, stream>>>(ipw, ipwb, (long long)DPROJ*DMODEL);
  k_cvt<<<(int)(((size_t)DMODEL*DINNER)/2048), blk, 0, stream>>>(opw, opwb, (long long)DMODEL*DINNER);

  // in_proj: zx[M, 4384] = u . ipw^T
  k_gemm_bf16<<<dim3(M/128, (DPROJ + 127)/128), blk, 0, stream>>>(
      ubf, ipwb, zx, M, DPROJ, DMODEL);

  for (int d = 0; d < 2; ++d) {
    const float* cw = d ? cwB  : cwF;
    const float* cb = d ? cbB  : cbF;
    const float* db = d ? dtbB : dtbF;
    const float* al = d ? alB  : alF;
    const float* Dd = d ? DB   : DF;
    const float* nw = d ? nwB  : nwF;

    k_conv  <<<(BSZ*LSEQ*CONVDIM)/256, blk, 0, stream>>>(zx, cw, cb, xbcb[d], d);
    k_dtcs  <<<BSZ*NHEADS*NCHUNK, blk, 0, stream>>>(zx, db, al, dtb[d], acsb[d], d);
    k_states<<<BSZ*NHEADS*NCHUNK, blk, 0, stream>>>(xbcb[d], dtb[d], acsb[d], st[d]);
    k_recur <<<BSZ*NHEADS,        blk, 0, stream>>>(st[d], acsb[d], pvb[d]);
    k_scan  <<<BSZ*NHEADS*NCHUNK, blk, 0, stream>>>(xbcb[d], dtb[d], acsb[d], pvb[d], Dd, yr[d]);
    k_gnorm <<<BSZ*LSEQ,          blk, 0, stream>>>(zx, nw, yr[d], d);
  }

  // bi-dir merge to bf16, then out_proj
  k_combine<<<(int)(((size_t)M*DINNER)/2048), blk, 0, stream>>>(yr[0], yr[1], ycb);
  k_gemm_bf16<<<dim3(M/128, DMODEL/128), blk, 0, stream>>>(
      ycb, opwb, out, M, DMODEL, DINNER);
}
